// MASTER_16492674417239
// MI455X (gfx1250) — matrix operand order repaired and checked by reading, not compiled, not run on hardware
//
#include <hip/hip_runtime.h>

// ---------------------------------------------------------------------------
// Types
// ---------------------------------------------------------------------------
typedef _Float16 half_t;
typedef __attribute__((ext_vector_type(16))) _Float16 v16h;
typedef __attribute__((ext_vector_type(8)))  _Float16 v8h;
typedef __attribute__((ext_vector_type(8)))  float    v8f;
typedef __attribute__((ext_vector_type(4)))  float    v4f;
typedef __attribute__((ext_vector_type(4)))  unsigned int v4u;
typedef __attribute__((ext_vector_type(8)))  int      v8i;
typedef __attribute__((ext_vector_type(4)))  int      v4i;

#define S_DIM  1024
#define T_DIM  64
#define F_DIM  128
#define J_DIM  64
#define E_DIM  256
#define H_DIM  4
#define DH_DIM 64
#define ROWS   (S_DIM * T_DIM)   // 65536 token rows

// Tensor Data Mover availability (arity differs between toolchains).
#if defined(__has_builtin)
#  if __has_builtin(__builtin_amdgcn_tensor_load_to_lds)
#    define USE_TDM 1
#  endif
#endif
#ifndef USE_TDM
#  define USE_TDM 0
#endif

// ---------------------------------------------------------------------------
// WMMA helpers (gfx1250 wave32, 16x16x32 f16 -> f32)
// ---------------------------------------------------------------------------
static __device__ inline v8f wmma_f16(v16h a, v16h b, v8f c) {
  return __builtin_amdgcn_wmma_f32_16x16x32_f16(false, a, false, b, (short)0, c,
                                                false, false);
}

// A fragment: lane L holds row (rbase + L%16); lanes 0-15 cover K = 0..7 and
// 16..23, lanes 16-31 cover K = 8..15 and 24..31 (per ISA 16-bit A layout).
static __device__ inline v16h load_frag_a(const half_t* lds, int stride,
                                          int rbase, int kbase, int lane) {
  int row = rbase + (lane & 15);
  int hl  = lane >> 4;
  const half_t* p = lds + row * stride + kbase + hl * 8;
  v8h lo = *(const v8h*)p;        // K = hl*8 + 0..7
  v8h hi = *(const v8h*)(p + 16); // K = 16 + hl*8 + 0..7
  return __builtin_shufflevector(lo, hi, 0, 1, 2, 3, 4, 5, 6, 7,
                                 8, 9, 10, 11, 12, 13, 14, 15);
}

// B fragment: LDS holds B transposed, Bt[n][k]. Lane L holds col (nbase+L%16);
// lanes 0-15 cover K=0..7 and 16..23, lanes 16-31 cover K=8..15 and 24..31.
static __device__ inline v16h load_frag_b(const half_t* lds, int stride,
                                          int nbase, int kbase, int lane) {
  int n  = nbase + (lane & 15);
  int hl = lane >> 4;
  const half_t* p = lds + n * stride + kbase + hl * 8;
  v8h lo = *(const v8h*)p;
  v8h hi = *(const v8h*)(p + 16);
  return __builtin_shufflevector(lo, hi, 0, 1, 2, 3, 4, 5, 6, 7,
                                 8, 9, 10, 11, 12, 13, 14, 15);
}

// ---------------------------------------------------------------------------
// Tensor Data Mover: DMA a 2D f16 tile (tile_w x tile_h, row stride = K elems)
// from global memory into LDS (contiguous tile_w-element rows).
// D# bitfields per CDNA5 ISA 8.3/8.4.  Issued once per wave (EXEC ignored).
// ---------------------------------------------------------------------------
#if USE_TDM
static __device__ inline void tdm_load_tile_f16(unsigned lds_off,
                                                const half_t* gptr,
                                                int tensor_w, int tensor_h,
                                                int tile_w, int tile_h,
                                                int row_stride) {
  unsigned long long ga = (unsigned long long)(uintptr_t)gptr;
  v4u g0;
  g0.x = 1u;                                        // count=1, user D#
  g0.y = lds_off;                                   // LDS byte address
  g0.z = (unsigned)(ga & 0xffffffffu);              // global_addr[31:0]
  g0.w = (unsigned)((ga >> 32) & 0x01ffffffu)       // global_addr[56:32]
         | 0x80000000u;                             // type = 2 ("image")
  v8i g1;
  g1[0] = (int)(1u << 16);                          // data_size=1 (2B), no flags
  g1[1] = (int)(((unsigned)tensor_w & 0xffffu) << 16);          // dim0[15:0]
  g1[2] = (int)((((unsigned)tensor_w >> 16) & 0xffffu)
         | (((unsigned)tensor_h & 0xffffu) << 16)); // dim0[31:16] | dim1[15:0]
  g1[3] = (int)((((unsigned)tensor_h >> 16) & 0xffffu)
         | (((unsigned)tile_w & 0xffffu) << 16));   // dim1[31:16] | tile_dim0
  g1[4] = (int)((unsigned)tile_h & 0xffffu);        // tile_dim1 (tile_dim2=0)
  g1[5] = row_stride;                               // tensor_dim0_stride[31:0]
  g1[6] = 0;
  g1[7] = 0;
  v4i gz = {0, 0, 0, 0};
#if defined(__clang_major__) && (__clang_major__ >= 23)
  v8i gz8 = {0, 0, 0, 0, 0, 0, 0, 0};
  __builtin_amdgcn_tensor_load_to_lds(g0, g1, gz, gz, gz8, 0);
#else
  __builtin_amdgcn_tensor_load_to_lds(g0, g1, gz, gz, 0);
#endif
}
#endif

// ---------------------------------------------------------------------------
// Generic WMMA GEMM:  C(MxN) = A_f16(MxK) * B_f16(KxN) [+bias][+resid][ReLU]
// Block tile 128x128, BK=32, 256 threads = 8 waves in 4x2; wave tile 32x64
// (2x4 WMMA C-tiles -> 8 WMMAs per 6 fragment loads per k-step).
// A tile staged by the Tensor Data Mover (wave 0), B staged transposed by ALU.
// ---------------------------------------------------------------------------
template <bool BIAS, bool RESID, bool RELU, bool OUTF, bool OUTH>
__global__ __launch_bounds__(256) void gemm_wmma_kernel(
    const half_t* __restrict__ A, const half_t* __restrict__ B,
    const float* __restrict__ bias, const float* __restrict__ resid,
    float* __restrict__ Cf, half_t* __restrict__ Ch,
    int M, int N, int K) {
  __shared__ half_t As[128 * 32];
  __shared__ half_t Bs[128 * 32];   // transposed: Bs[n][k]

  const int tid  = threadIdx.x;
  const int lane = tid & 31;
  const int wave = tid >> 5;
  const int m0 = blockIdx.y * 128;
  const int n0 = blockIdx.x * 128;
  const int wm = (wave >> 1) * 32;
  const int wn = (wave & 1) * 64;

  v8f acc[2][4] = {};

  for (int kt = 0; kt < K; kt += 32) {
    // ---- stage A tile (128 rows x 32 halves) ----
#if USE_TDM
    if (wave == 0) {
      tdm_load_tile_f16((unsigned)(uintptr_t)As,
                        A + (size_t)m0 * K + kt,
                        K, M, /*tile_w=*/32, /*tile_h=*/128, /*stride=*/K);
    }
#else
    for (int i = 0; i < 2; ++i) {
      int c   = tid * 2 + i;
      int row = c >> 2;
      int seg = (c & 3) * 8;
      *(v8h*)&As[row * 32 + seg] =
          *(const v8h*)&A[(size_t)(m0 + row) * K + kt + seg];
    }
#endif
    // ---- stage B tile transposed (32 k x 128 n -> Bs[n][k]) ----
    for (int i = 0; i < 2; ++i) {
      int c    = tid * 2 + i;
      int k    = c >> 4;
      int nseg = (c & 15) * 8;
      v8h bv = *(const v8h*)&B[(size_t)(kt + k) * N + n0 + nseg];
      for (int j = 0; j < 8; ++j) Bs[(nseg + j) * 32 + k] = bv[j];
    }
    if (kt + 32 < K) {
      __builtin_prefetch(&B[(size_t)(kt + 32 + (tid >> 4)) * N + n0], 0, 0);
#if !USE_TDM
      __builtin_prefetch(&A[(size_t)(m0 + (tid >> 1)) * K + kt + 32], 0, 0);
#endif
    }
#if USE_TDM
    if (wave == 0) __builtin_amdgcn_s_wait_tensorcnt(0);
#endif
    __syncthreads();

    v16h a0 = load_frag_a(As, 32, wm, 0, lane);
    v16h a1 = load_frag_a(As, 32, wm + 16, 0, lane);
    v16h b0 = load_frag_b(Bs, 32, wn, 0, lane);
    v16h b1 = load_frag_b(Bs, 32, wn + 16, 0, lane);
    v16h b2 = load_frag_b(Bs, 32, wn + 32, 0, lane);
    v16h b3 = load_frag_b(Bs, 32, wn + 48, 0, lane);
    acc[0][0] = wmma_f16(a0, b0, acc[0][0]);
    acc[0][1] = wmma_f16(a0, b1, acc[0][1]);
    acc[0][2] = wmma_f16(a0, b2, acc[0][2]);
    acc[0][3] = wmma_f16(a0, b3, acc[0][3]);
    acc[1][0] = wmma_f16(a1, b0, acc[1][0]);
    acc[1][1] = wmma_f16(a1, b1, acc[1][1]);
    acc[1][2] = wmma_f16(a1, b2, acc[1][2]);
    acc[1][3] = wmma_f16(a1, b3, acc[1][3]);
    __syncthreads();
  }

  const int hl = lane >> 4, ln16 = lane & 15;
  for (int ti = 0; ti < 2; ++ti) {
    for (int tj = 0; tj < 4; ++tj) {
      int gn = n0 + wn + tj * 16 + ln16;
      float bv = BIAS ? bias[gn] : 0.0f;
      for (int i = 0; i < 8; ++i) {
        int gm = m0 + wm + ti * 16 + i + hl * 8;   // C layout: VGPR i -> M=i/i+8
        size_t idx = (size_t)gm * N + gn;
        float val = acc[ti][tj][i] + bv;
        if (RESID) val += resid[idx];
        if (RELU) val = fmaxf(val, 0.0f);
        if (OUTF) Cf[idx] = val;
        if (OUTH) Ch[idx] = (half_t)val;
      }
    }
  }
}

// ---------------------------------------------------------------------------
// Flash attention (no score scale; output scaled by 1/sqrt(DH)=0.125 to match
// the reference).  Element (b,l,h,d) lives at base[b*sB + l*sL + h*64 + d].
// Grid: (L/64, H, nB).  64-row Q tile, 64-col KV tiles, online softmax.
// ---------------------------------------------------------------------------
__global__ __launch_bounds__(256) void flash_attn_kernel(
    const float* __restrict__ q, const float* __restrict__ k,
    const float* __restrict__ v, float* __restrict__ out,
    int L, int sB, int sL) {
  __shared__ half_t Qs[64 * 64];
  __shared__ half_t Ks[64 * 64];   // row-major [kv][d] == Bt with n=kv,k=d
  __shared__ half_t Vs[64 * 64];   // transposed  [d][kv] == Bt with n=d,k=kv
  __shared__ half_t Ps[64 * 64];
  __shared__ float  Ss[64 * 64];
  __shared__ float  mrow[64], lrow[64], srow[64];

  const int tid  = threadIdx.x;
  const int lane = tid & 31;
  const int wave = tid >> 5;
  const int h  = blockIdx.y;
  const int bz = blockIdx.z;
  const int q0 = blockIdx.x * 64;
  const size_t base = (size_t)bz * sB + (size_t)h * 64;

  // Stage Q (f32 -> f16), 16 elems/thread.
  {
    int l = tid >> 2, dseg = (tid & 3) * 16;
    const float* src = q + base + (size_t)(q0 + l) * sL + dseg;
    half_t* dst = &Qs[l * 64 + dseg];
    for (int i = 0; i < 16; i += 4) {
      v4f f = *(const v4f*)(src + i);
      dst[i] = (half_t)f[0]; dst[i + 1] = (half_t)f[1];
      dst[i + 2] = (half_t)f[2]; dst[i + 3] = (half_t)f[3];
    }
  }
  if (tid < 64) { mrow[tid] = -1e30f; lrow[tid] = 0.0f; }

  const int mt  = wave >> 1;        // 0..3  (q-row tile)
  const int ntb = (wave & 1) * 2;   // 0 or 2 (col tile base)
  const int hl = lane >> 4, ln16 = lane & 15;
  v8f o[2] = {};
  __syncthreads();

  for (int kv0 = 0; kv0 < L; kv0 += 64) {
    // Stage K (as-is) and V (transposed).
    {
      int l = tid >> 2, dseg = (tid & 3) * 16;
      const float* ksrc = k + base + (size_t)(kv0 + l) * sL + dseg;
      half_t* kdst = &Ks[l * 64 + dseg];
      for (int i = 0; i < 16; i += 4) {
        v4f f = *(const v4f*)(ksrc + i);
        kdst[i] = (half_t)f[0]; kdst[i + 1] = (half_t)f[1];
        kdst[i + 2] = (half_t)f[2]; kdst[i + 3] = (half_t)f[3];
      }
      const float* vsrc = v + base + (size_t)(kv0 + l) * sL + dseg;
      for (int i = 0; i < 16; i += 4) {
        v4f f = *(const v4f*)(vsrc + i);
        for (int j = 0; j < 4; ++j) Vs[(dseg + i + j) * 64 + l] = (half_t)f[j];
      }
    }
    __syncthreads();

    // S = Q * K^T  (64x64), each wave owns two 16x16 tiles.
    v8f s[2] = {};
    for (int kk = 0; kk < 64; kk += 32) {
      v16h a  = load_frag_a(Qs, 64, mt * 16, kk, lane);
      v16h b0 = load_frag_b(Ks, 64, ntb * 16, kk, lane);
      v16h b1 = load_frag_b(Ks, 64, (ntb + 1) * 16, kk, lane);
      s[0] = wmma_f16(a, b0, s[0]);
      s[1] = wmma_f16(a, b1, s[1]);
    }
    for (int j = 0; j < 2; ++j)
      for (int i = 0; i < 8; ++i)
        Ss[(mt * 16 + i + hl * 8) * 64 + (ntb + j) * 16 + ln16] = s[j][i];
    __syncthreads();

    // Online softmax: one thread per query row.
    if (tid < 64) {
      float mold = mrow[tid];
      float mmax = mold;
      const float* sr = &Ss[tid * 64];
      for (int j = 0; j < 64; ++j) mmax = fmaxf(mmax, sr[j]);
      float alpha = __expf(mold - mmax);
      float sum = 0.0f;
      half_t* pr = &Ps[tid * 64];
      for (int j = 0; j < 64; ++j) {
        float p = __expf(sr[j] - mmax);
        pr[j] = (half_t)p;
        sum += p;
      }
      lrow[tid] = lrow[tid] * alpha + sum;
      mrow[tid] = mmax;
      srow[tid] = alpha;
    }
    __syncthreads();

    // Rescale O accumulators, then O += P * V.
    for (int j = 0; j < 2; ++j)
      for (int i = 0; i < 8; ++i)
        o[j][i] *= srow[mt * 16 + i + hl * 8];
    for (int kk = 0; kk < 64; kk += 32) {
      v16h a  = load_frag_a(Ps, 64, mt * 16, kk, lane);
      v16h b0 = load_frag_b(Vs, 64, ntb * 16, kk, lane);
      v16h b1 = load_frag_b(Vs, 64, (ntb + 1) * 16, kk, lane);
      o[0] = wmma_f16(a, b0, o[0]);
      o[1] = wmma_f16(a, b1, o[1]);
    }
    __syncthreads();
  }

  // Finalize: divide by softmax denom, apply 1/sqrt(DH), write (also performs
  // the head-interleave and any (b,l) transpose via the strided indexing).
  for (int j = 0; j < 2; ++j) {
    for (int i = 0; i < 8; ++i) {
      int row = mt * 16 + i + hl * 8;
      int col = (ntb + j) * 16 + ln16;
      float val = o[j][i] / lrow[row] * 0.125f;
      out[base + (size_t)(q0 + row) * sL + col] = val;
    }
  }
}

// ---------------------------------------------------------------------------
// LayerNorm over E=256: one wave per row (8 rows/block).
// y = (x [+resid] [+posenc] - mean)/sqrt(var+eps)*g + b
// NOTE: resid/base_out may alias (in-place residual update) -> no restrict.
// ---------------------------------------------------------------------------
__global__ __launch_bounds__(256) void ln_kernel(
    const float* __restrict__ x, const float* resid,
    const float* __restrict__ g, const float* __restrict__ bta,
    float* base_out, float* __restrict__ ln_f32,
    half_t* __restrict__ ln_f16, int add_pos) {
  const int tid  = threadIdx.x;
  const int lane = tid & 31;
  const int wave = tid >> 5;
  const int r = blockIdx.x * 8 + wave;

  const float* xr = x + (size_t)r * E_DIM + lane * 8;
  float vals[8];
  v4f x0 = *(const v4f*)xr;
  v4f x1 = *(const v4f*)(xr + 4);
  vals[0] = x0[0]; vals[1] = x0[1]; vals[2] = x0[2]; vals[3] = x0[3];
  vals[4] = x1[0]; vals[5] = x1[1]; vals[6] = x1[2]; vals[7] = x1[3];
  if (resid) {
    const float* rr = resid + (size_t)r * E_DIM + lane * 8;
    v4f r0 = *(const v4f*)rr;
    v4f r1 = *(const v4f*)(rr + 4);
    vals[0] += r0[0]; vals[1] += r0[1]; vals[2] += r0[2]; vals[3] += r0[3];
    vals[4] += r1[0]; vals[5] += r1[1]; vals[6] += r1[2]; vals[7] += r1[3];
  }
  if (add_pos) {
    int t = r & (T_DIM - 1);
    for (int i = 0; i < 8; ++i) {
      int e = lane * 8 + i;
      float two_i = (float)(e & ~1);
      float div = __powf(100004.0f, two_i * (1.0f / (float)E_DIM));
      float pv = (float)t / div;
      vals[i] += (e & 1) ? __cosf(pv) : __sinf(pv);
    }
  }
  if (base_out) {
    float* br = base_out + (size_t)r * E_DIM + lane * 8;
    v4f b0 = {vals[0], vals[1], vals[2], vals[3]};
    v4f b1 = {vals[4], vals[5], vals[6], vals[7]};
    *(v4f*)br = b0;
    *(v4f*)(br + 4) = b1;
  }
  float s = 0.0f;
  for (int i = 0; i < 8; ++i) s += vals[i];
  for (int m = 16; m > 0; m >>= 1) s += __shfl_xor(s, m, 32);
  float mean = s * (1.0f / (float)E_DIM);
  float qv = 0.0f;
  for (int i = 0; i < 8; ++i) {
    float d = vals[i] - mean;
    qv += d * d;
  }
  for (int m = 16; m > 0; m >>= 1) qv += __shfl_xor(qv, m, 32);
  float rstd = rsqrtf(qv * (1.0f / (float)E_DIM) + 1e-10f);
  for (int i = 0; i < 8; ++i) {
    int e = lane * 8 + i;
    float y = (vals[i] - mean) * rstd * g[e] + bta[e];
    if (ln_f32) ln_f32[(size_t)r * E_DIM + e] = y;
    if (ln_f16) ln_f16[(size_t)r * E_DIM + e] = (half_t)y;
  }
}

// ---------------------------------------------------------------------------
// Gate: logits = x[:, -1, F:] @ gate_W + gate_b; softmax; gx = gate * x_f (f16)
// One block per sequence s, 128 threads (one per feature f).
// ---------------------------------------------------------------------------
__global__ __launch_bounds__(128) void gate_kernel(
    const float* __restrict__ x, const float* __restrict__ gW,
    const float* __restrict__ gb, half_t* __restrict__ gx) {
  const int s = blockIdx.x, f = threadIdx.x;
  __shared__ float red[128];
  const float* xg = x + ((size_t)s * T_DIM + (T_DIM - 1)) * (F_DIM + J_DIM) + F_DIM;
  float acc = gb[f];
  for (int j = 0; j < J_DIM; ++j) acc += xg[j] * gW[j * F_DIM + f];
  red[f] = acc;
  __syncthreads();
  for (int off = 64; off > 0; off >>= 1) {
    if (f < off) red[f] = fmaxf(red[f], red[f + off]);
    __syncthreads();
  }
  float mx = red[0];
  __syncthreads();
  float e = __expf(acc - mx);
  red[f] = e;
  __syncthreads();
  for (int off = 64; off > 0; off >>= 1) {
    if (f < off) red[f] += red[f + off];
    __syncthreads();
  }
  float gval = e / red[0];
  for (int t = 0; t < T_DIM; ++t) {
    float xf = x[((size_t)s * T_DIM + t) * (F_DIM + J_DIM) + f];
    gx[((size_t)s * T_DIM + t) * F_DIM + f] = (half_t)(gval * xf);
  }
}

// ---------------------------------------------------------------------------
// Final aggregation: lam = softmax_t(x1[s,t,:] . x1[s,-1,:]);
// agg = sum_t lam[t]*inter_out[s,t,:]; out[s] = agg . pred_W + pred_b
// ---------------------------------------------------------------------------
__global__ __launch_bounds__(256) void final_kernel(
    const float* __restrict__ x1, const float* __restrict__ io,
    const float* __restrict__ predW, const float* __restrict__ predb,
    float* __restrict__ out) {
  const int s = blockIdx.x, tid = threadIdx.x;
  const int lane = tid & 31, wave = tid >> 5;
  __shared__ float cur[E_DIM];
  __shared__ float lam[T_DIM];
  __shared__ float red[8];
  const float* xs = x1 + (size_t)s * T_DIM * E_DIM;
  cur[tid] = xs[(T_DIM - 1) * E_DIM + tid];
  __syncthreads();
  for (int i = 0; i < 8; ++i) {
    int t = wave * 8 + i;
    const float* xr = xs + t * E_DIM;
    float p = 0.0f;
    for (int j = lane; j < E_DIM; j += 32) p += xr[j] * cur[j];
    for (int m = 16; m > 0; m >>= 1) p += __shfl_xor(p, m, 32);
    if (lane == 0) lam[t] = p;
  }
  __syncthreads();
  if (tid == 0) {
    float mx = -1e30f;
    for (int t = 0; t < T_DIM; ++t) mx = fmaxf(mx, lam[t]);
    float sm = 0.0f;
    for (int t = 0; t < T_DIM; ++t) {
      float e = __expf(lam[t] - mx);
      lam[t] = e;
      sm += e;
    }
    float inv = 1.0f / sm;
    for (int t = 0; t < T_DIM; ++t) lam[t] *= inv;
  }
  __syncthreads();
  const float* ios = io + (size_t)s * T_DIM * E_DIM;
  float a = 0.0f;
  for (int t = 0; t < T_DIM; ++t) a += lam[t] * ios[t * E_DIM + tid];
  float p = a * predW[tid];
  for (int m = 16; m > 0; m >>= 1) p += __shfl_xor(p, m, 32);
  if (lane == 0) red[wave] = p;
  __syncthreads();
  if (tid == 0) {
    float ssum = 0.0f;
    for (int w = 0; w < 8; ++w) ssum += red[w];
    out[s] = ssum + predb[0];
  }
}

// ---------------------------------------------------------------------------
// Utilities: f32 -> f16 copy; QKV weight repack (H,E,DH) -> (E, H*DH) f16.
// ---------------------------------------------------------------------------
__global__ void f32_to_f16_kernel(const float* __restrict__ in,
                                  half_t* __restrict__ out, size_t n) {
  size_t i = (size_t)blockIdx.x * blockDim.x + threadIdx.x;
  if (i < n) out[i] = (half_t)in[i];
}

__global__ void repack_qkv_kernel(const float* __restrict__ W,
                                  half_t* __restrict__ out) {
  int i = blockIdx.x * 256 + threadIdx.x;  // 65536 total
  int d = i & 63;
  int e = (i >> 6) & 255;
  int h = i >> 14;
  out[e * 256 + h * 64 + d] =
      (half_t)W[(size_t)h * (E_DIM * DH_DIM) + (size_t)e * DH_DIM + d];
}

// ---------------------------------------------------------------------------
// Host-side orchestration.
// ---------------------------------------------------------------------------
extern "C" void kernel_launch(void* const* d_in, const int* in_sizes, int n_in,
                              void* d_out, int out_size, void* d_ws,
                              size_t ws_size, hipStream_t stream) {
  (void)in_sizes; (void)n_in; (void)out_size; (void)ws_size;
  const float* x          = (const float*)d_in[0];
  const float* gate_W     = (const float*)d_in[1];
  const float* gate_b     = (const float*)d_in[2];
  const float* embed_W    = (const float*)d_in[3];
  const float* ln0_g      = (const float*)d_in[4];
  const float* ln0_b      = (const float*)d_in[5];
  const float* intra_Wq   = (const float*)d_in[6];
  const float* intra_Wk   = (const float*)d_in[7];
  const float* intra_Wv   = (const float*)d_in[8];
  const float* intra_ln_g = (const float*)d_in[9];
  const float* intra_ln_b = (const float*)d_in[10];
  const float* intra_W1   = (const float*)d_in[11];
  const float* intra_b1   = (const float*)d_in[12];
  const float* intra_W2   = (const float*)d_in[13];
  const float* intra_b2   = (const float*)d_in[14];
  const float* inter_Wq   = (const float*)d_in[15];
  const float* inter_Wk   = (const float*)d_in[16];
  const float* inter_Wv   = (const float*)d_in[17];
  const float* inter_ln1_g = (const float*)d_in[18];
  const float* inter_ln1_b = (const float*)d_in[19];
  const float* inter_ln2_g = (const float*)d_in[20];
  const float* inter_ln2_b = (const float*)d_in[21];
  const float* inter_W1   = (const float*)d_in[22];
  const float* inter_b1   = (const float*)d_in[23];
  const float* inter_W2   = (const float*)d_in[24];
  const float* inter_b2   = (const float*)d_in[25];
  const float* temp_W     = (const float*)d_in[26];
  const float* pred_W     = (const float*)d_in[27];
  const float* pred_b     = (const float*)d_in[28];
  float* out = (float*)d_out;

  char* ws = (char*)d_ws;
  size_t off = 0;
  auto alloc = [&](size_t bytes) -> char* {
    char* p = ws + off;
    off += (bytes + 255) & ~(size_t)255;
    return p;
  };
  half_t* gxA  = (half_t*)alloc((size_t)ROWS * F_DIM * sizeof(half_t));
  half_t* wEmb = (half_t*)alloc((size_t)F_DIM * E_DIM * sizeof(half_t));
  half_t* wQ   = (half_t*)alloc((size_t)E_DIM * E_DIM * sizeof(half_t));
  half_t* wK   = (half_t*)alloc((size_t)E_DIM * E_DIM * sizeof(half_t));
  half_t* wV   = (half_t*)alloc((size_t)E_DIM * E_DIM * sizeof(half_t));
  half_t* wQ2  = (half_t*)alloc((size_t)E_DIM * E_DIM * sizeof(half_t));
  half_t* wK2  = (half_t*)alloc((size_t)E_DIM * E_DIM * sizeof(half_t));
  half_t* wV2  = (half_t*)alloc((size_t)E_DIM * E_DIM * sizeof(half_t));
  half_t* wF1a = (half_t*)alloc((size_t)E_DIM * 4 * E_DIM * sizeof(half_t));
  half_t* wF2a = (half_t*)alloc((size_t)4 * E_DIM * E_DIM * sizeof(half_t));
  half_t* wF1b = (half_t*)alloc((size_t)E_DIM * 4 * E_DIM * sizeof(half_t));
  half_t* wF2b = (half_t*)alloc((size_t)4 * E_DIM * E_DIM * sizeof(half_t));
  half_t* wTmp = (half_t*)alloc((size_t)E_DIM * E_DIM * sizeof(half_t));
  float*  BUF1 = (float*)alloc((size_t)ROWS * E_DIM * sizeof(float));   // ln0 -> r -> a2 -> r2
  half_t* BUF2 = (half_t*)alloc((size_t)ROWS * E_DIM * sizeof(half_t)); // f16 activations
  float*  Qb   = (float*)alloc((size_t)ROWS * E_DIM * sizeof(float));
  float*  Kb   = (float*)alloc((size_t)ROWS * E_DIM * sizeof(float));
  float*  Vb   = (float*)alloc((size_t)ROWS * E_DIM * sizeof(float));
  float*  BUF6 = (float*)alloc((size_t)ROWS * E_DIM * sizeof(float));   // emb -> a -> intra_out -> inter_out
  half_t* Hb   = (half_t*)alloc((size_t)16384 * 1024 * sizeof(half_t)); // FFN hidden chunk

  const dim3 blk256(256);

  // ---- weight preprocessing (f32 -> f16 / repack) ----
  f32_to_f16_kernel<<<(F_DIM * E_DIM) / 256, blk256, 0, stream>>>(embed_W, wEmb, (size_t)F_DIM * E_DIM);
  f32_to_f16_kernel<<<(E_DIM * 4 * E_DIM) / 256, blk256, 0, stream>>>(intra_W1, wF1a, (size_t)E_DIM * 4 * E_DIM);
  f32_to_f16_kernel<<<(4 * E_DIM * E_DIM) / 256, blk256, 0, stream>>>(intra_W2, wF2a, (size_t)4 * E_DIM * E_DIM);
  f32_to_f16_kernel<<<(E_DIM * 4 * E_DIM) / 256, blk256, 0, stream>>>(inter_W1, wF1b, (size_t)E_DIM * 4 * E_DIM);
  f32_to_f16_kernel<<<(4 * E_DIM * E_DIM) / 256, blk256, 0, stream>>>(inter_W2, wF2b, (size_t)4 * E_DIM * E_DIM);
  f32_to_f16_kernel<<<(E_DIM * E_DIM) / 256, blk256, 0, stream>>>(temp_W, wTmp, (size_t)E_DIM * E_DIM);
  repack_qkv_kernel<<<256, blk256, 0, stream>>>(intra_Wq, wQ);
  repack_qkv_kernel<<<256, blk256, 0, stream>>>(intra_Wk, wK);
  repack_qkv_kernel<<<256, blk256, 0, stream>>>(intra_Wv, wV);
  repack_qkv_kernel<<<256, blk256, 0, stream>>>(inter_Wq, wQ2);
  repack_qkv_kernel<<<256, blk256, 0, stream>>>(inter_Wk, wK2);
  repack_qkv_kernel<<<256, blk256, 0, stream>>>(inter_Wv, wV2);

  // ---- gate + gated input (f16) ----
  gate_kernel<<<S_DIM, dim3(128), 0, stream>>>(x, gate_W, gate_b, gxA);

  // ---- embedding GEMM: (65536x128)x(128x256) -> BUF6 ----
  gemm_wmma_kernel<false, false, false, true, false>
      <<<dim3(E_DIM / 128, ROWS / 128), blk256, 0, stream>>>(
          gxA, wEmb, nullptr, nullptr, BUF6, nullptr, ROWS, E_DIM, F_DIM);

  // ---- +posenc, LN0 -> BUF1 (f32 residual) + BUF2 (f16 GEMM A) ----
  ln_kernel<<<ROWS / 8, blk256, 0, stream>>>(BUF6, nullptr, ln0_g, ln0_b,
                                             nullptr, BUF1, BUF2, 1);

  // ---- intra QKV ----
  gemm_wmma_kernel<false, false, false, true, false>
      <<<dim3(2, ROWS / 128), blk256, 0, stream>>>(
          BUF2, wQ, nullptr, nullptr, Qb, nullptr, ROWS, E_DIM, E_DIM);
  gemm_wmma_kernel<false, false, false, true, false>
      <<<dim3(2, ROWS / 128), blk256, 0, stream>>>(
          BUF2, wK, nullptr, nullptr, Kb, nullptr, ROWS, E_DIM, E_DIM);
  gemm_wmma_kernel<false, false, false, true, false>
      <<<dim3(2, ROWS / 128), blk256, 0, stream>>>(
          BUF2, wV, nullptr, nullptr, Vb, nullptr, ROWS, E_DIM, E_DIM);

  // ---- intra attention: batch=s (stride 64*256), length=T ----
  flash_attn_kernel<<<dim3(1, H_DIM, S_DIM), blk256, 0, stream>>>(
      Qb, Kb, Vb, BUF6, T_DIM, T_DIM * E_DIM, E_DIM);

  // ---- r = a + ln0 (into BUF1), LN(r) f16 -> BUF2 ----
  ln_kernel<<<ROWS / 8, blk256, 0, stream>>>(BUF6, BUF1, intra_ln_g, intra_ln_b,
                                             BUF1, nullptr, BUF2, 0);

  // ---- intra FFN (chunked over rows): intra_out -> BUF6 ----
  for (int c = 0; c < 4; ++c) {
    size_t r0 = (size_t)c * 16384;
    gemm_wmma_kernel<true, false, true, false, true>
        <<<dim3(1024 / 128, 16384 / 128), blk256, 0, stream>>>(
            BUF2 + r0 * E_DIM, wF1a, intra_b1, nullptr, nullptr, Hb,
            16384, 1024, E_DIM);
    gemm_wmma_kernel<true, true, false, true, false>
        <<<dim3(E_DIM / 128, 16384 / 128), blk256, 0, stream>>>(
            Hb, wF2a, intra_b2, BUF1 + r0 * E_DIM, BUF6 + r0 * E_DIM, nullptr,
            16384, E_DIM, 1024);
  }

  // ---- inter: LN1 f16 -> BUF2, QKV, flash attention over S per (h,t) ----
  ln_kernel<<<ROWS / 8, blk256, 0, stream>>>(BUF6, nullptr, inter_ln1_g,
                                             inter_ln1_b, nullptr, nullptr,
                                             BUF2, 0);
  gemm_wmma_kernel<false, false, false, true, false>
      <<<dim3(2, ROWS / 128), blk256, 0, stream>>>(
          BUF2, wQ2, nullptr, nullptr, Qb, nullptr, ROWS, E_DIM, E_DIM);
  gemm_wmma_kernel<false, false, false, true, false>
      <<<dim3(2, ROWS / 128), blk256, 0, stream>>>(
          BUF2, wK2, nullptr, nullptr, Kb, nullptr, ROWS, E_DIM, E_DIM);
  gemm_wmma_kernel<false, false, false, true, false>
      <<<dim3(2, ROWS / 128), blk256, 0, stream>>>(
          BUF2, wV2, nullptr, nullptr, Vb, nullptr, ROWS, E_DIM, E_DIM);
  // batch=t (stride 256), length=S (stride 64*256); output lands in (s,t,e).
  flash_attn_kernel<<<dim3(S_DIM / 64, H_DIM, T_DIM), blk256, 0, stream>>>(
      Qb, Kb, Vb, BUF1, S_DIM, E_DIM, T_DIM * E_DIM);

  // ---- r2 = a2 + intra_out (into BUF1), LN2 f16 -> BUF2 ----
  ln_kernel<<<ROWS / 8, blk256, 0, stream>>>(BUF1, BUF6, inter_ln2_g,
                                             inter_ln2_b, BUF1, nullptr, BUF2, 0);

  // ---- inter FFN (chunked): inter_out -> BUF6 ----
  for (int c = 0; c < 4; ++c) {
    size_t r0 = (size_t)c * 16384;
    gemm_wmma_kernel<true, false, true, false, true>
        <<<dim3(1024 / 128, 16384 / 128), blk256, 0, stream>>>(
            BUF2 + r0 * E_DIM, wF1b, inter_b1, nullptr, nullptr, Hb,
            16384, 1024, E_DIM);
    gemm_wmma_kernel<true, true, false, true, false>
        <<<dim3(E_DIM / 128, 16384 / 128), blk256, 0, stream>>>(
            Hb, wF2b, inter_b2, BUF1 + r0 * E_DIM, BUF6 + r0 * E_DIM, nullptr,
            16384, E_DIM, 1024);
  }

  // ---- x1 = inter_out @ temp_W -> Qb ----
  f32_to_f16_kernel<<<(ROWS * E_DIM) / 256, blk256, 0, stream>>>(
      BUF6, BUF2, (size_t)ROWS * E_DIM);
  gemm_wmma_kernel<false, false, false, true, false>
      <<<dim3(2, ROWS / 128), blk256, 0, stream>>>(
          BUF2, wTmp, nullptr, nullptr, Qb, nullptr, ROWS, E_DIM, E_DIM);

  // ---- lambda softmax + aggregation + prediction ----
  final_kernel<<<S_DIM, blk256, 0, stream>>>(Qb, BUF6, pred_W, pred_b, out);
}
